// CustomGIN_46033459478730
// MI455X (gfx1250) — matrix operand order repaired and checked by reading, not compiled, not run on hardware
//
#include <hip/hip_runtime.h>

typedef __attribute__((ext_vector_type(16))) _Float16 v16h;
typedef __attribute__((ext_vector_type(8)))  _Float16 v8h;
typedef __attribute__((ext_vector_type(8)))  float    v8f;

typedef unsigned int u32x4 __attribute__((ext_vector_type(4)));
typedef int          i32x4 __attribute__((ext_vector_type(4)));
typedef int          i32x8 __attribute__((ext_vector_type(8)));

#define GIN_HID     128
#define GIN_GRAPHS  64
#define GIN_LAYERS  3

// ---------------------------------------------------------------------------
// Weight prep: WT[m][n][k] = (f16) W[m][k][n]   (row n = output column,
// contiguous in k so WMMA B-fragments are two contiguous 8-half LDS reads)
// ---------------------------------------------------------------------------
__global__ void gin_prep_w(const float* __restrict__ W, _Float16* __restrict__ WT,
                           int nmat) {
    int i = blockIdx.x * 256 + threadIdx.x;
    if (i >= nmat * 16384) return;
    int m = i >> 14;
    int n = (i >> 7) & 127;
    int k = i & 127;
    WT[i] = (_Float16)W[(size_t)m * 16384 + (size_t)k * 128 + n];
}

// ---------------------------------------------------------------------------
// Y = act(X @ W + b), X:[n_rows,128] f32, WT:[128,128] f16 (pre-transposed),
// per-block 128-row tile, 8 waves, v_wmma_f32_16x16x32_f16.
// Y has padded rows (n_rows rounded up to 128) -> unconditional stores.
// Weight tile staged LDS<-global by the Tensor Data Mover (one wave issues
// tensor_load_to_lds, waits on TENSORcnt, block barriers).
// ---------------------------------------------------------------------------
template <int RELU>
__global__ __launch_bounds__(256)
void gin_gemm(const float* __restrict__ X, const _Float16* __restrict__ WT,
              const float* __restrict__ bias, float* __restrict__ Y,
              int n_rows) {
    __shared__ _Float16 Xs[128 * 128];   // 32 KB: activation tile (f16)
    __shared__ _Float16 Ws[128 * 128];   // 32 KB: W^T (f16)

    const int tid  = threadIdx.x;
    const int row0 = blockIdx.x * 128;

    // ---- stage W^T via TDM (wave 0 only; EXEC/divergence irrelevant to TDM)
    if (tid < 32) {
#if defined(__HIP_DEVICE_COMPILE__) && __has_builtin(__builtin_amdgcn_tensor_load_to_lds) && __has_builtin(__builtin_amdgcn_s_wait_tensorcnt)
        const unsigned lds_off = (unsigned)(unsigned long long)(void*)&Ws[0];
        const unsigned long long ga = (unsigned long long)(const void*)WT;
        // D# group0: count=1 | lds_addr | global_addr | type=2
        u32x4 g0 = {1u, lds_off, (unsigned)ga,
                    (unsigned)((ga >> 32) & 0x1FFFFFFull) | (2u << 30)};
        // D# group1: data_size=2B, tensor 128x128, tile 128x128, stride0=128
        i32x8 g1 = {(int)(1u << 16),     // data_size=1 (2 bytes), wg_mask=0
                    (int)(128u << 16),   // tensor_dim0 low16 in [63:48]
                    (int)(128u << 16),   // tensor_dim1 low16 in [95:80]
                    (int)(128u << 16),   // tile_dim0   in [127:112]
                    128,                 // tile_dim1   in [143:128]
                    128,                 // tensor_dim0_stride low32
                    0, 0};
        i32x4 gz = {0, 0, 0, 0};
#if __clang_major__ >= 23
        i32x8 gz8 = {0, 0, 0, 0, 0, 0, 0, 0};
        __builtin_amdgcn_tensor_load_to_lds(g0, g1, gz, gz, gz8, 0);
#else
        __builtin_amdgcn_tensor_load_to_lds(g0, g1, gz, gz, 0);
#endif
        __builtin_amdgcn_s_wait_tensorcnt(0);
#else
        // fallback: plain wave-0 copy (32 KB)
        const uint4* src = (const uint4*)WT;
        uint4*       dst = (uint4*)Ws;
        for (int i = tid; i < 2048; i += 32) dst[i] = src[i];
#endif
    }

    // ---- stage X tile: f32 -> f16, row-major (all 8 waves)
    for (int i = tid; i < 128 * 32; i += 256) {
        int r  = i >> 5;
        int c4 = (i & 31) << 2;
        int gr = row0 + r;
        float4 v = {0.f, 0.f, 0.f, 0.f};
        if (gr < n_rows) v = ((const float4*)X)[(size_t)gr * 32 + (i & 31)];
        _Float16* p = &Xs[r * 128 + c4];
        p[0] = (_Float16)v.x; p[1] = (_Float16)v.y;
        p[2] = (_Float16)v.z; p[3] = (_Float16)v.w;
    }
    __syncthreads();

    const int wave = tid >> 5;
    const int lane = tid & 31;
    const int lhi  = lane >> 4;    // half-wave select
    const int llo  = lane & 15;

    // Hoist the 4 A-fragments (16x32 f16 each) for this wave's 16-row tile.
    // ISA layout: lanes 0-15 M=lane, K = kb..kb+7 then kb+16..kb+23,
    //             lanes 16-31 same M with kb += 8.
    v16h a[4];
    {
        const int arow = (wave * 16 + llo) * 128;
#pragma unroll
        for (int kt = 0; kt < 4; ++kt) {
            const int kb = kt * 32 + (lhi ? 8 : 0);
            v8h lo = *(const v8h*)&Xs[arow + kb];
            v8h hi = *(const v8h*)&Xs[arow + kb + 16];
            v16h av;
#pragma unroll
            for (int j = 0; j < 8; ++j) { av[j] = lo[j]; av[8 + j] = hi[j]; }
            a[kt] = av;
        }
    }

#pragma unroll
    for (int nt = 0; nt < 8; ++nt) {
        const float bv = bias[nt * 16 + llo];
        v8f acc;
#pragma unroll
        for (int r = 0; r < 8; ++r) acc[r] = bv;   // fold bias into accumulator

        // B layout: lane holds column n = nt*16+llo, K = kb..kb+7 then kb+16..kb+23
        // in W^T row n; lanes 16-31 use kb += 8.
        const int brow = (nt * 16 + llo) * 128;
#pragma unroll
        for (int kt = 0; kt < 4; ++kt) {
            const int kb = kt * 32 + (lhi ? 8 : 0);
            v8h blo = *(const v8h*)&Ws[brow + kb];
            v8h bhi = *(const v8h*)&Ws[brow + kb + 16];
            v16h b;
#pragma unroll
            for (int j = 0; j < 8; ++j) { b[j] = blo[j]; b[8 + j] = bhi[j]; }
            acc = __builtin_amdgcn_wmma_f32_16x16x32_f16(
                false, a[kt], false, b, (short)0, acc, false, false);
        }

        // D layout: element (m = r + 8*lhi, n = llo) in VGPR r.
        // Y rows are padded to the block tile -> no bounds check, no branches.
        float* yp = Y + (size_t)(row0 + wave * 16 + lhi * 8) * 128 + nt * 16 + llo;
#pragma unroll
        for (int r = 0; r < 8; ++r) {
            float v = acc[r];
            if (RELU) v = fmaxf(v, 0.f);
            yp[(size_t)r * 128] = v;
        }
    }
}

// ---------------------------------------------------------------------------
// z = (1 + eps[layer]) * h      (streaming, float4)
// ---------------------------------------------------------------------------
__global__ void gin_scale(const float* __restrict__ h, float* __restrict__ z,
                          const float* __restrict__ eps, int layer, int n4) {
    int i = blockIdx.x * 256 + threadIdx.x;
    if (i >= n4) return;
    const float s = 1.0f + eps[layer];
    float4 v = ((const float4*)h)[i];
    float4 o = {s * v.x, s * v.y, s * v.z, s * v.w};
    ((float4*)z)[i] = o;
}

// ---------------------------------------------------------------------------
// z[dst] += h[src] over edges: one wave per edge, lane owns 4 floats.
// Coalesced float4 gather + native f32 fadd atomics for the scatter.
// ---------------------------------------------------------------------------
__global__ __launch_bounds__(256)
void gin_edge_agg(const float* __restrict__ h, const long long* __restrict__ ei,
                  float* __restrict__ z, int n_edges) {
    const int wid  = (blockIdx.x * 256 + threadIdx.x) >> 5;
    const int lane = threadIdx.x & 31;
    if (wid >= n_edges) return;
    const size_t s = (size_t)ei[wid];
    const size_t d = (size_t)ei[(size_t)n_edges + wid];
    float4 v = ((const float4*)(h + s * 128))[lane];
    float* zp = z + d * 128 + (size_t)lane * 4;
    unsafeAtomicAdd(zp + 0, v.x);
    unsafeAtomicAdd(zp + 1, v.y);
    unsafeAtomicAdd(zp + 2, v.z);
    unsafeAtomicAdd(zp + 3, v.w);
}

// ---------------------------------------------------------------------------
// Per-graph sum pool; batch is sorted, so accumulate runs in registers and
// only touch atomics at segment boundaries. 128 threads = feature columns.
// ---------------------------------------------------------------------------
__global__ __launch_bounds__(128)
void gin_pool(const float* __restrict__ h, const long long* __restrict__ batch,
              float* __restrict__ out, int layer, int n_nodes) {
    __shared__ int gids[512];
    const int start = blockIdx.x * 512;
    const int cnt   = min(512, n_nodes - start);
    for (int i = threadIdx.x; i < cnt; i += 128) gids[i] = (int)batch[start + i];
    __syncthreads();

    const int t = threadIdx.x;
    float acc = 0.f;
    int   cur = gids[0];
    for (int i = 0; i < cnt; ++i) {
        const int g = gids[i];                 // uniform across block
        if (g != cur) {
            unsafeAtomicAdd(&out[cur * (GIN_LAYERS * GIN_HID) + layer * GIN_HID + t], acc);
            acc = 0.f;
            cur = g;
        }
        acc += h[(size_t)(start + i) * 128 + t];
    }
    unsafeAtomicAdd(&out[cur * (GIN_LAYERS * GIN_HID) + layer * GIN_HID + t], acc);
}

// ---------------------------------------------------------------------------
extern "C" void kernel_launch(void* const* d_in, const int* in_sizes, int n_in,
                              void* d_out, int out_size, void* d_ws, size_t ws_size,
                              hipStream_t stream) {
    const float*     x      = (const float*)d_in[0];
    const long long* ei     = (const long long*)d_in[1];
    const long long* bat    = (const long long*)d_in[2];
    const float*     W_pre1 = (const float*)d_in[3];
    const float*     b_pre1 = (const float*)d_in[4];
    const float*     W_pre2 = (const float*)d_in[5];
    const float*     b_pre2 = (const float*)d_in[6];
    const float*     W1s    = (const float*)d_in[7];
    const float*     b1s    = (const float*)d_in[8];
    const float*     W2s    = (const float*)d_in[9];
    const float*     b2s    = (const float*)d_in[10];
    const float*     eps    = (const float*)d_in[11];

    const int n_nodes = in_sizes[0] / 128;
    const int n_edges = in_sizes[1] / 2;
    const int n_pad   = (n_nodes + 127) & ~127;   // GEMM outputs padded to tile

    // workspace layout: [8 x 128x128 f16 weights][h][mid][z] (padded rows)
    _Float16* wt  = (_Float16*)d_ws;
    float* hbuf = (float*)((char*)d_ws + (size_t)8 * 16384 * sizeof(_Float16));
    float* mbuf = hbuf + (size_t)n_pad * 128;
    float* zbuf = mbuf + (size_t)n_pad * 128;

    // transpose + f16-convert all weight matrices
    gin_prep_w<<<(16384 + 255) / 256, 256, 0, stream>>>(W_pre1, wt, 1);
    gin_prep_w<<<(16384 + 255) / 256, 256, 0, stream>>>(W_pre2, wt + 16384, 1);
    gin_prep_w<<<(3 * 16384 + 255) / 256, 256, 0, stream>>>(W1s, wt + 2 * 16384, 3);
    gin_prep_w<<<(3 * 16384 + 255) / 256, 256, 0, stream>>>(W2s, wt + 5 * 16384, 3);

    hipMemsetAsync(d_out, 0,
                   (size_t)GIN_GRAPHS * GIN_LAYERS * GIN_HID * sizeof(float), stream);

    const int gblocks = n_pad / 128;
    const int n4      = (n_nodes * 128) / 4;
    const int sblocks = (n4 + 255) / 256;
    const int eblocks = (n_edges + 7) / 8;
    const int pblocks = (n_nodes + 511) / 512;

    // pre-MLP: h = relu(relu(x @ Wp1 + bp1) @ Wp2 + bp2)   (h lives in mbuf)
    gin_gemm<1><<<gblocks, 256, 0, stream>>>(x, wt, b_pre1, hbuf, n_nodes);
    gin_gemm<1><<<gblocks, 256, 0, stream>>>(hbuf, wt + 16384, b_pre2, mbuf, n_nodes);

    float* h = mbuf;      // current node features
    for (int l = 0; l < GIN_LAYERS; ++l) {
        gin_scale<<<sblocks, 256, 0, stream>>>(h, zbuf, eps, l, n4);
        gin_edge_agg<<<eblocks, 256, 0, stream>>>(h, ei, zbuf, n_edges);
        gin_gemm<1><<<gblocks, 256, 0, stream>>>(zbuf, wt + (2 + l) * 16384,
                                                 b1s + l * 128, hbuf, n_nodes);
        gin_gemm<0><<<gblocks, 256, 0, stream>>>(hbuf, wt + (5 + l) * 16384,
                                                 b2s + l * 128, h, n_nodes);
        gin_pool<<<pblocks, 128, 0, stream>>>(h, bat, (float*)d_out, l, n_nodes);
    }
}
